// SlidingWindowAttentionBlock_678604833213
// MI455X (gfx1250) — matrix operand order repaired and checked by reading, not compiled, not run on hardware
//
#include <hip/hip_runtime.h>
#include <hip/hip_bf16.h>
#include <math.h>

// ---------------------------------------------------------------------------
// Sliding-window attention transformer block for MI455X (gfx1250, wave32).
//   B=2, D=64, H=W=128, heads=4 (hd=16), window K=7, FFN=4.
// GEMMs: v_wmma_f32_16x16x32_f16, register-blocked 16x64 per wave
// (1 A fragment reused against 4 B fragments per k-step).
// ---------------------------------------------------------------------------

typedef _Float16 v16h __attribute__((ext_vector_type(16)));
typedef _Float16 v8h  __attribute__((ext_vector_type(8)));
typedef float    v8f  __attribute__((ext_vector_type(8)));

#define NPIX   32768            // B*H*W = 2*128*128
#define HWSZ   16384            // H*W
#define MTILES 2048             // NPIX / 16

union V16U { v16h v; v8h h[2]; };

struct h2pair { _Float16 x, y; };

// A fragment: 16x32 f16, row-major source, per ISA 7.12.2 layout.
// lane<16: M=lane, K chunks {0..7, 16..23}; lane>=16: same M, K {8..15, 24..31}.
__device__ __forceinline__ v16h load_a_frag(const _Float16* __restrict__ A,
                                            int lda, int kbase, int lane) {
  const int m  = lane & 15;
  const int lh = lane >> 4;
  const _Float16* row = A + (size_t)m * lda + kbase + lh * 8;
  V16U u;
  u.h[0] = *(const v8h*)(row);        // K = kbase + lh*8      .. +7
  u.h[1] = *(const v8h*)(row + 16);   // K = kbase + 16 + lh*8 .. +7
  return u.v;
}

// B fragment: 32x16 f16 where B[k][n] = W[n0+n][k] (out = act @ W.T).
// lane holds column n = lane&15; K chunks {kbase+(lane>>4)*8 .. +7, kbase+16+(lane>>4)*8 .. +7} in W row.
__device__ __forceinline__ v16h load_b_frag(const _Float16* __restrict__ W,
                                            int ldw, int n0, int kbase, int lane) {
  const int n  = lane & 15;
  const int lh = lane >> 4;
  const _Float16* row = W + (size_t)(n0 + n) * ldw + kbase + lh * 8;
  V16U u;
  u.h[0] = *(const v8h*)(row);
  u.h[1] = *(const v8h*)(row + 16);
  return u.v;
}

// Register-blocked GEMM core: one wave produces a 16 x (16*NT) tile.
// A fragment loaded once per k-step and reused against NT B fragments.
template<int KDIM, int NT>
__device__ __forceinline__ void gemm_tile(const _Float16* __restrict__ Atile,
                                          int lda,
                                          const _Float16* __restrict__ W,
                                          int ldw, int n0, int lane,
                                          v8f (&acc)[NT]) {
  #pragma unroll
  for (int kb = 0; kb < KDIM; kb += 32) {
    v16h a = load_a_frag(Atile, lda, kb, lane);
    #pragma unroll
    for (int t = 0; t < NT; ++t) {
      v16h b = load_b_frag(W, ldw, n0 + 16 * t, kb, lane);
      acc[t] = __builtin_amdgcn_wmma_f32_16x16x32_f16(false, a, false, b,
                                                      (short)0, acc[t],
                                                      false, false);
    }
  }
}

__device__ __forceinline__ int refl(int i, int n) {
  i = (i < 0) ? -i : i;
  i = (i >= n) ? (2 * n - 2 - i) : i;
  return i;
}

// --------------------------- kernel 1: weight repack -----------------------
__global__ void prep_weights_kernel(const float* __restrict__ wq,
                                    const float* __restrict__ wk,
                                    const float* __restrict__ wv,
                                    const float* __restrict__ wo,
                                    const float* __restrict__ w1,
                                    const float* __restrict__ w2,
                                    _Float16* __restrict__ wqkv_h,
                                    _Float16* __restrict__ wo_h,
                                    _Float16* __restrict__ w1_h,
                                    _Float16* __restrict__ w2_h) {
  int i = blockIdx.x * blockDim.x + threadIdx.x;
  if (i < 4096)        wqkv_h[i]          = (_Float16)wq[i];
  else if (i < 8192)   wqkv_h[i]          = (_Float16)wk[i - 4096];
  else if (i < 12288)  wqkv_h[i]          = (_Float16)wv[i - 8192];
  else if (i < 16384)  wo_h[i - 12288]    = (_Float16)wo[i - 12288];
  else if (i < 32768)  w1_h[i - 16384]    = (_Float16)w1[i - 16384];
  else if (i < 49152)  w2_h[i - 32768]    = (_Float16)w2[i - 32768];
}

// --------------------------- kernel 2: LN1 ---------------------------------
// x is [B,64,128,128]; channel stride = HWSZ, so for fixed c adjacent threads
// (adjacent pixels) are fully coalesced.
__global__ void ln1_kernel(const float* __restrict__ x,
                           const float* __restrict__ g,
                           const float* __restrict__ bt,
                           _Float16* __restrict__ xln) {
  int p = blockIdx.x * blockDim.x + threadIdx.x;
  if (p >= NPIX) return;
  int b = p >> 14, hw = p & (HWSZ - 1);
  const float* base = x + (size_t)b * 64 * HWSZ + hw;
  float s = 0.f, s2 = 0.f;
  #pragma unroll 8
  for (int c = 0; c < 64; ++c) {
    float v = base[(size_t)c * HWSZ];
    s += v; s2 += v * v;
  }
  float mean = s * (1.0f / 64.0f);
  float var  = s2 * (1.0f / 64.0f) - mean * mean;
  float rstd = rsqrtf(var + 1e-5f);
  _Float16* dst = xln + (size_t)p * 64;
  #pragma unroll 8
  for (int c = 0; c < 64; ++c) {
    float v = base[(size_t)c * HWSZ];
    dst[c] = (_Float16)((v - mean) * rstd * g[c] + bt[c]);
  }
}

// --------------------------- kernel 3: QKV GEMM ----------------------------
// [NPIX,64] x [64,192] -> qkv f16 [NPIX,192] (q:0-63, k:64-127, v:128-191)
// grid (MTILES/4, 3): blockIdx.y picks q/k/v 64-col group; wave -> pixel tile.
__global__ void qkv_gemm_kernel(const _Float16* __restrict__ xln,
                                const _Float16* __restrict__ wqkv,
                                _Float16* __restrict__ qkv) {
  const int lane = threadIdx.x & 31;
  const int wave = threadIdx.x >> 5;
  const int m0   = (blockIdx.x * 4 + wave) * 16;
  const int n0   = blockIdx.y * 64;
  v8f acc[4] = {};
  gemm_tile<64, 4>(xln + (size_t)m0 * 64, 64, wqkv, 64, n0, lane, acc);
  const int n = lane & 15, lh = lane >> 4;
  #pragma unroll
  for (int t = 0; t < 4; ++t) {
    #pragma unroll
    for (int j = 0; j < 8; ++j) {
      int m = m0 + lh * 8 + j;
      qkv[(size_t)m * 192 + n0 + 16 * t + n] = (_Float16)acc[t][j];
    }
  }
}

// --------------------------- kernel 4: windowed attention ------------------
// One wave per pixel; lane holds 2 channels; head = lane>>3.
// Online softmax; score reduced over the 8-lane head group with shfl_xor.
__global__ void attn_kernel(const _Float16* __restrict__ qkv,
                            _Float16* __restrict__ ao) {
  const int wave = threadIdx.x >> 5;
  const int lane = threadIdx.x & 31;
  const int p    = blockIdx.x * 8 + wave;
  const int b  = p >> 14;
  const int hw = p & (HWSZ - 1);
  const int hh = hw >> 7, ww = hw & 127;
  const int ch = lane * 2;

  h2pair qp = *(const h2pair*)(qkv + (size_t)p * 192 + ch);
  const float q0 = (float)qp.x, q1 = (float)qp.y;

  float mrun = -3.0e38f, ssum = 0.f, a0 = 0.f, a1 = 0.f;
  #pragma unroll 7
  for (int idx = 0; idx < 49; ++idx) {
    int dy = idx / 7 - 3, dx = idx % 7 - 3;
    int ny = refl(hh + dy, 128);
    int nx = refl(ww + dx, 128);
    size_t np = (size_t)b * HWSZ + ny * 128 + nx;
    const _Float16* kb = qkv + np * 192 + 64 + ch;
    h2pair kp = *(const h2pair*)kb;
    h2pair vp = *(const h2pair*)(kb + 64);
    float part = q0 * (float)kp.x + q1 * (float)kp.y;
    part += __shfl_xor(part, 1, 32);
    part += __shfl_xor(part, 2, 32);
    part += __shfl_xor(part, 4, 32);       // head-group (8 lanes) reduction
    float s    = part * 0.25f;             // 1/sqrt(hd=16)
    float mnew = fmaxf(mrun, s);
    float corr = expf(mrun - mnew);
    float wgt  = expf(s - mnew);
    ssum = ssum * corr + wgt;
    a0   = a0 * corr + wgt * (float)vp.x;
    a1   = a1 * corr + wgt * (float)vp.y;
    mrun = mnew;
  }
  float inv = 1.0f / ssum;
  ao[(size_t)p * 64 + ch]     = (_Float16)(a0 * inv);
  ao[(size_t)p * 64 + ch + 1] = (_Float16)(a1 * inv);
}

// ---------------- kernel 5: Wo GEMM + residual + LN2 (fused via LDS) -------
// grid MTILES/4, block 128: each wave computes 16 pixels x all 64 cols;
// block covers 64 pixels -> LN2 over the 64x64 LDS tile (2 threads/pixel).
__global__ void wo_ln2_kernel(const _Float16* __restrict__ ao,
                              const _Float16* __restrict__ wo_h,
                              const float* __restrict__ x,
                              const float* __restrict__ g2,
                              const float* __restrict__ b2v,
                              float* __restrict__ y,
                              _Float16* __restrict__ hln) {
  __shared__ float sy[64][64];
  const int lane = threadIdx.x & 31;
  const int wave = threadIdx.x >> 5;
  const int mb   = blockIdx.x * 64;           // block pixel base
  const int m0   = mb + wave * 16;            // wave pixel tile
  v8f acc[4] = {};
  gemm_tile<64, 4>(ao + (size_t)m0 * 64, 64, wo_h, 64, 0, lane, acc);
  const int n = lane & 15, lh = lane >> 4;
  #pragma unroll
  for (int t = 0; t < 4; ++t) {
    const int col = 16 * t + n;
    #pragma unroll
    for (int j = 0; j < 8; ++j) {
      int m  = m0 + lh * 8 + j;
      int b_ = m >> 14, hw = m & (HWSZ - 1);
      float val = acc[t][j] + x[((size_t)b_ * 64 + col) * HWSZ + hw]; // residual
      y[(size_t)m * 64 + col] = val;
      sy[m - mb][col] = val;
    }
  }
  __syncthreads();
  // LN2: 2 threads per pixel (lane pairs), 32 channels each.
  const int px  = threadIdx.x >> 1;     // 0..63
  const int sub = threadIdx.x & 1;      // channels sub*32..+31
  float s = 0.f, s2 = 0.f;
  #pragma unroll
  for (int i = 0; i < 32; ++i) {
    float v = sy[px][sub * 32 + i];
    s += v; s2 += v * v;
  }
  s  += __shfl_xor(s, 1, 32);
  s2 += __shfl_xor(s2, 1, 32);
  float mean = s * (1.0f / 64.0f);
  float var  = s2 * (1.0f / 64.0f) - mean * mean;
  float rstd = rsqrtf(var + 1e-5f);
  int pg = mb + px;
  #pragma unroll
  for (int i = 0; i < 32; ++i) {
    int cc = sub * 32 + i;
    float v = (sy[px][cc] - mean) * rstd * g2[cc] + b2v[cc];
    hln[(size_t)pg * 64 + cc] = (_Float16)v;
  }
}

// --------------------------- kernel 6: FFN1 + GELU -------------------------
// grid (MTILES/4, 4): blockIdx.y picks a 64-col group of the 256 outputs.
__global__ void ffn1_kernel(const _Float16* __restrict__ hln,
                            const _Float16* __restrict__ w1_h,
                            const float* __restrict__ b1,
                            _Float16* __restrict__ gbuf) {
  const int lane = threadIdx.x & 31;
  const int wave = threadIdx.x >> 5;
  const int m0   = (blockIdx.x * 4 + wave) * 16;
  const int n0   = blockIdx.y * 64;
  v8f acc[4] = {};
  gemm_tile<64, 4>(hln + (size_t)m0 * 64, 64, w1_h, 64, n0, lane, acc);
  const int n = lane & 15, lh = lane >> 4;
  #pragma unroll
  for (int t = 0; t < 4; ++t) {
    const int col  = n0 + 16 * t + n;
    const float bias = b1[col];
    #pragma unroll
    for (int j = 0; j < 8; ++j) {
      int m = m0 + lh * 8 + j;
      float v  = acc[t][j] + bias;
      float ge = 0.5f * v * (1.0f + erff(v * 0.70710678f));   // exact GELU
      gbuf[(size_t)m * 256 + col] = (_Float16)ge;
    }
  }
}

// --------------------------- kernel 7: FFN2 + residual + transpose ---------
// K=256 -> 8 k-steps x 4 N tiles = 32 WMMAs per wave.
__global__ void ffn2_kernel(const _Float16* __restrict__ gbuf,
                            const _Float16* __restrict__ w2_h,
                            const float* __restrict__ b2,
                            const float* __restrict__ y,
                            float* __restrict__ out) {
  const int lane = threadIdx.x & 31;
  const int wave = threadIdx.x >> 5;
  const int m0   = (blockIdx.x * 4 + wave) * 16;
  v8f acc[4] = {};
  gemm_tile<256, 4>(gbuf + (size_t)m0 * 256, 256, w2_h, 256, 0, lane, acc);
  const int n = lane & 15, lh = lane >> 4;
  #pragma unroll
  for (int t = 0; t < 4; ++t) {
    const int col  = 16 * t + n;
    const float bias = b2[col];
    #pragma unroll
    for (int j = 0; j < 8; ++j) {
      int m  = m0 + lh * 8 + j;
      int b_ = m >> 14, hw = m & (HWSZ - 1);
      float v = acc[t][j] + bias + y[(size_t)m * 64 + col];
      out[((size_t)b_ * 64 + col) * HWSZ + hw] = v;   // [B,D,H,W]
    }
  }
}

// ---------------------------------------------------------------------------
extern "C" void kernel_launch(void* const* d_in, const int* in_sizes, int n_in,
                              void* d_out, int out_size, void* d_ws, size_t ws_size,
                              hipStream_t stream) {
  (void)in_sizes; (void)n_in; (void)out_size; (void)ws_size;
  const float* x     = (const float*)d_in[0];
  const float* ln1g  = (const float*)d_in[1];
  const float* ln1b  = (const float*)d_in[2];
  const float* Wq    = (const float*)d_in[3];
  const float* Wk    = (const float*)d_in[4];
  const float* Wv    = (const float*)d_in[5];
  const float* Wo    = (const float*)d_in[6];
  const float* ln2g  = (const float*)d_in[7];
  const float* ln2b  = (const float*)d_in[8];
  const float* W1    = (const float*)d_in[9];
  const float* b1    = (const float*)d_in[10];
  const float* W2    = (const float*)d_in[11];
  const float* b2    = (const float*)d_in[12];
  float* out = (float*)d_out;

  // Workspace layout (all chunk sizes multiples of 256 B; ~50.4 MB total,
  // fits comfortably inside the 192 MB L2 so intermediates stay on-chip).
  char* w = (char*)d_ws;
  _Float16* wqkv_h = (_Float16*)(w);                       // 192*64*2   = 24576
  _Float16* wo_h   = (_Float16*)(w + 24576);               // 64*64*2    = 8192
  _Float16* w1_h   = (_Float16*)(w + 32768);               // 256*64*2   = 32768
  _Float16* w2_h   = (_Float16*)(w + 65536);               // 64*256*2   = 32768
  _Float16* xln    = (_Float16*)(w + 98304);               // 32768*64*2 = 4 MiB
  _Float16* qkv    = (_Float16*)(w + 98304 + (4u<<20));    // 32768*192*2= 12 MiB
  _Float16* ao     = (_Float16*)(w + 98304 + (16u<<20));   // 4 MiB
  float*    y      = (float*)   (w + 98304 + (20u<<20));   // 32768*64*4 = 8 MiB
  _Float16* hln    = (_Float16*)(w + 98304 + (28u<<20));   // 4 MiB
  _Float16* gbuf   = (_Float16*)(w + 98304 + (32u<<20));   // 32768*256*2= 16 MiB

  prep_weights_kernel<<<192, 256, 0, stream>>>(Wq, Wk, Wv, Wo, W1, W2,
                                               wqkv_h, wo_h, w1_h, w2_h);
  ln1_kernel<<<NPIX / 256, 256, 0, stream>>>(x, ln1g, ln1b, xln);
  qkv_gemm_kernel<<<dim3(MTILES / 4, 3), 128, 0, stream>>>(xln, wqkv_h, qkv);
  attn_kernel<<<NPIX / 8, 256, 0, stream>>>(qkv, ao);
  wo_ln2_kernel<<<MTILES / 4, 128, 0, stream>>>(ao, wo_h, x, ln2g, ln2b, y, hln);
  ffn1_kernel<<<dim3(MTILES / 4, 4), 128, 0, stream>>>(hln, w1_h, b1, gbuf);
  ffn2_kernel<<<MTILES / 4, 128, 0, stream>>>(gbuf, w2_h, b2, y, out);
}
